// MultiHeadAttention_23201413333588
// MI455X (gfx1250) — compile-verified, matrix operand order checked by reading, not run on hardware
//
#include <hip/hip_runtime.h>
#include <cstdint>

// MultiHeadAttention forward, CDNA5 (gfx1250) bf16-WMMA implementation.
// out = (softmax(mask(QWq*s . (KWk)^T)) . (VWv)) Wo ; plus head-mean attn map.

#define D_MODEL 1024
#define N_HEAD  16
#define DH      64
#define BATCH   8
#define SEQ     1024
#define MTOT    (BATCH * SEQ)   // 8192 rows for all projection GEMMs
#define SC_STRIDE 1032          // padded f32 row stride (LDS bank spread)
#define PB_STRIDE 1040          // padded bf16 row stride

typedef __attribute__((ext_vector_type(16))) __bf16 v16bf;
typedef __attribute__((ext_vector_type(8)))  float  v8f;

union Frag16 { v16bf v; unsigned int u[8]; };

__device__ __forceinline__ unsigned short f2bf(float f) {
  unsigned int u = __float_as_uint(f);
  u += 0x7FFFu + ((u >> 16) & 1u);       // round-to-nearest-even
  return (unsigned short)(u >> 16);
}

// ---------------------------------------------------------------------------
// Y = (X[M,1024] @ W[1024,1024] + bias) * scale
// Block: 256 thr (8 waves), tile 128 rows x 64 cols, K-step 32 (one wmma-K).
// IN_BF16: X tile staged with global_load_async_to_lds_b128 (ASYNCcnt path).
// OUT_VT : store transposed per batch as Y[b][col][seq] (bf16) so attention
//          context B-fragments get contiguous key-pairs.
// ---------------------------------------------------------------------------
template<bool IN_BF16, bool OUT_F32, bool OUT_VT>
__global__ __launch_bounds__(256) void gemm_bias_kernel(
    const void* __restrict__ Xv, const float* __restrict__ W,
    const float* __restrict__ bias, void* __restrict__ Yv, float scale)
{
  __shared__ unsigned short Xs[128][34];   // bf16 X tile, +2 pad (17 dwords/row)
  __shared__ unsigned short Ws[64][34];    // bf16 W tile, stored transposed [col][k]

  const int tid  = threadIdx.x;
  const int lane = tid & 31;
  const int wave = tid >> 5;
  const int n0   = blockIdx.x * 64;
  const int row0 = blockIdx.y * 128;
  const int m    = lane & 15;              // M (or N) index within 16
  const int half = lane >> 4;

  v8f acc[4];
  v8f zf = {};
  #pragma unroll
  for (int j = 0; j < 4; ++j) acc[j] = zf;

  for (int k0 = 0; k0 < D_MODEL; k0 += 32) {
    if (IN_BF16) {
      // bf16 source: DMA 16B chunks straight into LDS, no VGPR round-trip.
      #pragma unroll
      for (int i = 0; i < 2; ++i) {
        int idx = i * 256 + tid;           // 512 chunks of 16B (128 rows x 4)
        int r = idx >> 2, ch = idx & 3;
        const unsigned short* src = (const unsigned short*)Xv +
            (size_t)(row0 + r) * D_MODEL + k0 + ch * 8;
        unsigned int ldsoff = (unsigned int)(uintptr_t)(&Xs[r][ch * 8]);
        asm volatile("global_load_async_to_lds_b128 %0, %1, off"
                     :: "v"(ldsoff), "v"(src) : "memory");
      }
    } else {
      #pragma unroll
      for (int i = 0; i < 16; ++i) {       // 128x32 X tile, f32->bf16
        int idx = i * 256 + tid;
        int r = idx >> 5, c = idx & 31;
        Xs[r][c] = f2bf(((const float*)Xv)[(size_t)(row0 + r) * D_MODEL + k0 + c]);
      }
      if (k0 + 32 < D_MODEL)               // global_prefetch_b8 of next tile
        __builtin_prefetch((const char*)Xv +
            ((size_t)(row0 + (tid & 127)) * D_MODEL + k0 + 32) * 4, 0, 1);
    }
    #pragma unroll
    for (int i = 0; i < 8; ++i) {          // 32x64 W tile, transposed into LDS
      int idx = i * 256 + tid;
      int kk = idx >> 6, c = idx & 63;
      Ws[c][kk] = f2bf(W[(size_t)(k0 + kk) * D_MODEL + n0 + c]);
    }
    if (k0 + 32 < D_MODEL)
      __builtin_prefetch((const char*)W +
          ((size_t)(k0 + 32 + (tid & 31)) * D_MODEL + n0) * 4, 0, 1);
    if (IN_BF16)
      asm volatile("s_wait_asynccnt 0" ::: "memory");
    __syncthreads();

    // A fragment: rows wave*16..+15, K pairs per ISA 16-bit 16x32 layout
    Frag16 a;
    #pragma unroll
    for (int j = 0; j < 8; ++j) {
      int kb = (j >> 2) * 16 + half * 8 + (j & 3) * 2;
      a.u[j] = *(const unsigned int*)&Xs[wave * 16 + m][kb];
    }
    #pragma unroll
    for (int nt = 0; nt < 4; ++nt) {
      Frag16 bfr;
      #pragma unroll
      for (int j = 0; j < 8; ++j) {
        int kb = (j >> 2) * 16 + half * 8 + (j & 3) * 2;
        bfr.u[j] = *(const unsigned int*)&Ws[nt * 16 + m][kb];
      }
      acc[nt] = __builtin_amdgcn_wmma_f32_16x16x32_bf16(
          false, a.v, false, bfr.v, (short)0, acc[nt], false, false);
    }
    __syncthreads();
  }

  #pragma unroll
  for (int nt = 0; nt < 4; ++nt) {
    #pragma unroll
    for (int r = 0; r < 8; ++r) {          // C layout: M = half*8 + r, N = lane&15
      int row = row0 + wave * 16 + half * 8 + r;
      int col = n0 + nt * 16 + m;
      float v = (acc[nt][r] + bias[col]) * scale;
      if (OUT_VT) {
        int bb = row >> 10, srow = row & (SEQ - 1);
        ((unsigned short*)Yv)[((size_t)bb * D_MODEL + col) * SEQ + srow] = f2bf(v);
      } else if (OUT_F32) {
        ((float*)Yv)[(size_t)row * D_MODEL + col] = v;
      } else {
        ((unsigned short*)Yv)[(size_t)row * D_MODEL + col] = f2bf(v);
      }
    }
  }
}

// ---------------------------------------------------------------------------
// Attention: one block per (batch b, 16-row query tile). Loops all 16 heads so
// the head-averaged attention map accumulates in REGISTERS (each softmax
// thread owns fixed (row, col-phase) cells) and the [B,H,S,S] tensor is never
// materialized. Mask is packed once per block into a 2 KB LDS bitfield via
// wave32 ballot.
// ---------------------------------------------------------------------------
__global__ __launch_bounds__(256) void attn_kernel(
    const unsigned short* __restrict__ Qp, const unsigned short* __restrict__ Kp,
    const unsigned short* __restrict__ Vt, const int* __restrict__ mask,
    unsigned short* __restrict__ ctxg, float* __restrict__ amean_out)
{
  extern __shared__ char smem_raw[];
  float*          sc   = (float*)smem_raw;                // [16][SC_STRIDE]
  float*          ctx  = sc  + 16 * SC_STRIDE;            // [16][64]
  float*          red  = ctx + 16 * 64;                   // [16][16]
  float*          rmx  = red + 256;                       // [16]
  float*          rsm  = rmx + 16;                        // [16]
  unsigned int*   mb   = (unsigned int*)(rsm + 16);       // [16][32] mask bits
  unsigned short* pbuf = (unsigned short*)(mb + 16 * 32); // [16][PB_STRIDE]

  const int tid  = threadIdx.x;
  const int lane = tid & 31;
  const int wave = tid >> 5;
  const int b    = blockIdx.x >> 6;
  const int q0   = (blockIdx.x & 63) * 16;
  const int m    = lane & 15;
  const int half = lane >> 4;
  const size_t bS = (size_t)b * SEQ;
  const float NEG_INF = -__builtin_huge_valf();

  // head-mean accumulators: thread (r=tid>>4, c16=tid&15) owns cols c16+16j
  float amv[64];
  #pragma unroll
  for (int j = 0; j < 64; ++j) amv[j] = 0.0f;

  // pack mask slice into bits, once per block
#if __has_builtin(__builtin_amdgcn_ballot_w32)
  for (int wi = wave; wi < 16 * 32; wi += 8) {
    int r = wi >> 5, w = wi & 31;
    int mval = mask[(bS + q0 + r) * SEQ + w * 32 + lane];
    unsigned int bits = __builtin_amdgcn_ballot_w32(mval != 0);
    if (lane == 0) mb[wi] = bits;
  }
#else
  for (int i = tid; i < 16 * 32; i += 256) {
    int r = i >> 5, w = i & 31;
    const int* mrow = mask + (bS + q0 + r) * SEQ + w * 32;
    unsigned int bits = 0;
    #pragma unroll
    for (int t = 0; t < 32; ++t) bits |= (mrow[t] != 0 ? 1u : 0u) << t;
    mb[i] = bits;
  }
#endif
  __syncthreads();

  for (int h = 0; h < N_HEAD; ++h) {
    const int hd = h * DH;

    // Q fragments (reused for 64 key tiles): two 16x32 bf16 A-frags (dh=64)
    Frag16 qa[2];
    #pragma unroll
    for (int c = 0; c < 2; ++c)
      #pragma unroll
      for (int j = 0; j < 8; ++j) {
        int kb = (j >> 2) * 16 + half * 8 + (j & 3) * 2;
        qa[c].u[j] = *(const unsigned int*)
            &Qp[(bS + q0 + m) * D_MODEL + hd + c * 32 + kb];
      }

    // ---- scores: wave w owns keys [w*128, w*128+128) ----
    for (int t = 0; t < 8; ++t) {
      int kbase = wave * 128 + t * 16;
      Frag16 kb0, kb1;                      // B-frags: N=key, K=dh (pairs contiguous)
      #pragma unroll
      for (int j = 0; j < 8; ++j) {
        int kb = (j >> 2) * 16 + half * 8 + (j & 3) * 2;
        kb0.u[j] = *(const unsigned int*)&Kp[(bS + kbase + m) * D_MODEL + hd + kb];
        kb1.u[j] = *(const unsigned int*)&Kp[(bS + kbase + m) * D_MODEL + hd + 32 + kb];
      }
      v8f s = {};
      s = __builtin_amdgcn_wmma_f32_16x16x32_bf16(false, qa[0].v, false, kb0.v, (short)0, s, false, false);
      s = __builtin_amdgcn_wmma_f32_16x16x32_bf16(false, qa[1].v, false, kb1.v, (short)0, s, false, false);
      const int key   = kbase + m;
      const int kword = key >> 5, kbit = key & 31;
      #pragma unroll
      for (int r = 0; r < 8; ++r) {
        int ml = half * 8 + r;
        bool msk = (mb[ml * 32 + kword] >> kbit) & 1u;   // nonzero mask -> -inf
        sc[ml * SC_STRIDE + key] = msk ? NEG_INF : s[r];
      }
    }
    __syncthreads();

    // ---- softmax: 16 threads per row, row = tid>>4 ----
    {
      int r = tid >> 4, c16 = tid & 15;
      float mx = NEG_INF;
      for (int j = 0; j < 64; ++j)
        mx = fmaxf(mx, sc[r * SC_STRIDE + c16 + j * 16]);
      red[r * 16 + c16] = mx;
      __syncthreads();
      if (c16 == 0) {
        float v = red[r * 16];
        for (int j = 1; j < 16; ++j) v = fmaxf(v, red[r * 16 + j]);
        rmx[r] = v;
      }
      __syncthreads();
      float rm = rmx[r], s = 0.0f;
      for (int j = 0; j < 64; ++j) {
        int c = c16 + j * 16;
        float e = __expf(sc[r * SC_STRIDE + c] - rm);
        sc[r * SC_STRIDE + c] = e;
        s += e;
      }
      red[r * 16 + c16] = s;
      __syncthreads();
      if (c16 == 0) {
        float v = 0.0f;
        for (int j = 0; j < 16; ++j) v += red[r * 16 + j];
        rsm[r] = v;
      }
      __syncthreads();
      float inv = 1.0f / rsm[r];
      #pragma unroll
      for (int j = 0; j < 64; ++j) {
        int c = c16 + j * 16;
        float p = sc[r * SC_STRIDE + c] * inv;
        amv[j] += p * (1.0f / N_HEAD);          // register head-mean accum
        pbuf[r * PB_STRIDE + c] = f2bf(p);      // bf16 P tile, converted once
      }
    }
    for (int i = tid; i < 16 * 64; i += 256) ctx[i] = 0.0f;
    __syncthreads();

    // ---- context: ctx[16x64] = P[16x1024] @ V[1024x64] ----
    {
      const int ntile = wave & 3;             // which 16-wide dh slice
      const int khalf = wave >> 2;            // which half of the 1024 keys
      const unsigned short* vrow =            // Vt row: contiguous along keys
          Vt + ((size_t)b * D_MODEL + hd + ntile * 16 + m) * SEQ;
      v8f cacc = {};
      for (int cc = 0; cc < 16; ++cc) {
        int kc = (khalf * 16 + cc) * 32;
        Frag16 pa, vb;
        #pragma unroll
        for (int j = 0; j < 8; ++j) {
          int kb = (j >> 2) * 16 + half * 8 + (j & 3) * 2;
          pa.u[j] = *(const unsigned int*)&pbuf[m * PB_STRIDE + kc + kb];
          vb.u[j] = *(const unsigned int*)&vrow[kc + kb];
        }
        cacc = __builtin_amdgcn_wmma_f32_16x16x32_bf16(
            false, pa.v, false, vb.v, (short)0, cacc, false, false);
      }
      #pragma unroll
      for (int r = 0; r < 8; ++r) {           // cross-wave combine via ds_add_f32
        int ml = half * 8 + r;
        atomicAdd(&ctx[ml * 64 + ntile * 16 + m], cacc[r]);
      }
    }
    __syncthreads();

    for (int i = tid; i < 16 * 64; i += 256) {
      int r = i >> 6, d = i & 63;
      ctxg[(bS + q0 + r) * D_MODEL + hd + d] = f2bf(ctx[r * 64 + d]);
    }
    __syncthreads();
  }

  // bounce register head-mean through (now dead) sc for a coalesced store
  {
    int r = tid >> 4, c16 = tid & 15;
    #pragma unroll
    for (int j = 0; j < 64; ++j)
      sc[r * SC_STRIDE + c16 + j * 16] = amv[j];
  }
  __syncthreads();
  for (int i = tid; i < 16 * 1024; i += 256) {
    int r = i >> 10, c = i & 1023;
    amean_out[(bS + q0 + r) * SEQ + c] = sc[r * SC_STRIDE + c];
  }
}

// ---------------------------------------------------------------------------
extern "C" void kernel_launch(void* const* d_in, const int* in_sizes, int n_in,
                              void* d_out, int out_size, void* d_ws, size_t ws_size,
                              hipStream_t stream) {
  (void)in_sizes; (void)n_in; (void)out_size; (void)ws_size;

  const float* k_in = (const float*)d_in[0];
  const float* v_in = (const float*)d_in[1];
  const float* q_in = (const float*)d_in[2];
  const int*   msk  = (const int*)  d_in[3];
  const float* Wk   = (const float*)d_in[4];
  const float* bk   = (const float*)d_in[5];
  const float* Wv   = (const float*)d_in[6];
  const float* bv   = (const float*)d_in[7];
  const float* Wq   = (const float*)d_in[8];
  const float* bq   = (const float*)d_in[9];
  const float* Wo   = (const float*)d_in[10];
  const float* bo   = (const float*)d_in[11];

  // workspace: bf16 Kp, Vt (transposed), Qp, context (16 MB each)
  unsigned short* Kp = (unsigned short*)d_ws;
  unsigned short* Vt = Kp + (size_t)MTOT * D_MODEL;
  unsigned short* Qp = Vt + (size_t)MTOT * D_MODEL;
  unsigned short* Cx = Qp + (size_t)MTOT * D_MODEL;

  float* out   = (float*)d_out;                         // [B,S,D]
  float* amean = out + (size_t)MTOT * D_MODEL;          // [B,S,S]

  dim3 ggrid(D_MODEL / 64, MTOT / 128);
  gemm_bias_kernel<false,false,false><<<ggrid, 256, 0, stream>>>(k_in, Wk, bk, Kp, 1.0f);
  gemm_bias_kernel<false,false,true ><<<ggrid, 256, 0, stream>>>(v_in, Wv, bv, Vt, 1.0f);
  gemm_bias_kernel<false,false,false><<<ggrid, 256, 0, stream>>>(q_in, Wq, bq, Qp, 0.125f); // 1/sqrt(64)

  const int smem = (16 * SC_STRIDE + 16 * 64 + 256 + 32) * (int)sizeof(float)
                 + 16 * 32 * (int)sizeof(unsigned int)
                 + 16 * PB_STRIDE * (int)sizeof(unsigned short);   // ~104 KB
  (void)hipFuncSetAttribute((const void*)attn_kernel,
                            hipFuncAttributeMaxDynamicSharedMemorySize, smem);
  attn_kernel<<<dim3(BATCH * (SEQ / 16)), 256, smem, stream>>>(Qp, Kp, Vt, msk, Cx, amean);

  gemm_bias_kernel<true,true,false><<<ggrid, 256, 0, stream>>>(Cx, Wo, bo, out, 1.0f);
}
